// Encoder_1331439861799
// MI455X (gfx1250) — matrix operand order repaired and checked by reading, not compiled, not run on hardware
//
#include <hip/hip_runtime.h>

#define N_NODES 100000
#define N_EDGES 640000
#define HID 128
#define OUTC 64
#define BATCH 1024

typedef __attribute__((ext_vector_type(16))) __bf16 v16bf;
typedef __attribute__((ext_vector_type(8)))  float  v8f;
typedef __attribute__((ext_vector_type(8)))  __bf16 bf8;
typedef __attribute__((ext_vector_type(4)))  __bf16 bf4;

__device__ __forceinline__ __bf16 f2bf(float f) {
  union { float f; unsigned u; } v; v.f = f;
  unsigned r = v.u + 0x7FFFu + ((v.u >> 16) & 1u);   // round-to-nearest-even
  unsigned short hs = (unsigned short)(r >> 16);
  __bf16 b; __builtin_memcpy(&b, &hs, 2); return b;
}
__device__ __forceinline__ float bf2f(__bf16 b) {
  unsigned short hs; __builtin_memcpy(&hs, &b, 2);
  union { unsigned u; float f; } v; v.u = ((unsigned)hs) << 16; return v.f;
}

// ---------------------------------------------------------------- zero init
__global__ void k_zero(float* p, size_t n) {
  size_t i = (size_t)blockIdx.x * blockDim.x + threadIdx.x;
  size_t stride = (size_t)gridDim.x * blockDim.x;
  for (; i < n; i += stride) p[i] = 0.0f;
}

// ------------------------------------------------- weights f32 -> bf16 once
__global__ void k_cvt_w(const float* Wl1, const float* Wr1,
                        const float* Wl2, const float* Wr2,
                        __bf16* Wl1b, __bf16* Wr1b, __bf16* Wl2b, __bf16* Wr2b) {
  int t = blockIdx.x * blockDim.x + threadIdx.x;
  if (t < 16384)       Wl1b[t]         = f2bf(Wl1[t]);
  else if (t < 32768)  Wr1b[t - 16384] = f2bf(Wr1[t - 16384]);
  else if (t < 40960)  Wl2b[t - 32768] = f2bf(Wl2[t - 32768]);
  else if (t < 49152)  Wr2b[t - 40960] = f2bf(Wr2[t - 40960]);
}

// ------------------------------------------- layer 0 edge scatter (4 feats)
__global__ void k_edge0(const long long* ei, const float* x,
                        float* deg, float* agg0) {
  int e = blockIdx.x * blockDim.x + threadIdx.x;
  if (e >= N_EDGES) return;
  int s = (int)ei[e];
  int d = (int)ei[N_EDGES + e];
  atomicAdd(&deg[d], 1.0f);
  const float4 xv = *(const float4*)(x + (size_t)s * 4);
  float* a = agg0 + (size_t)d * 4;
  atomicAdd(a + 0, xv.x);
  atomicAdd(a + 1, xv.y);
  atomicAdd(a + 2, xv.z);
  atomicAdd(a + 3, xv.w);
}

// ---- per-(node,feature): x_theta, layer-0 SAGE + PReLU, g1 = x1 + x_theta
__global__ void k_node0(const float* x, const float* deg, const float* agg0,
                        const float* W1, const float* Wl0, const float* bl0,
                        const float* Wr0, const float* a0,
                        float* x_theta, float* h, __bf16* g1b) {
  long long t = (long long)blockIdx.x * blockDim.x + threadIdx.x;
  if (t >= (long long)N_NODES * HID) return;
  int n = (int)(t >> 7);
  int f = (int)(t & 127);
  const float4 xv = *(const float4*)(x + (size_t)n * 4);
  float rd = 1.0f / fmaxf(deg[n], 1.0f);
  float4 mv = *(const float4*)(agg0 + (size_t)n * 4);
  mv.x *= rd; mv.y *= rd; mv.z *= rd; mv.w *= rd;
  const float4 w1 = *(const float4*)(W1 + f * 4);
  const float4 wl = *(const float4*)(Wl0 + f * 4);
  const float4 wr = *(const float4*)(Wr0 + f * 4);
  float xt = w1.x * xv.x + w1.y * xv.y + w1.z * xv.z + w1.w * xv.w;
  float v  = wl.x * mv.x + wl.y * mv.y + wl.z * mv.z + wl.w * mv.w + bl0[f]
           + wr.x * xv.x + wr.y * xv.y + wr.z * xv.z + wr.w * xv.w;
  float a  = a0[f];
  float x1 = (v >= 0.0f) ? v : a * v;
  x_theta[t] = xt;
  h[t]       = x1;
  g1b[t]     = f2bf(x1 + xt);
}

// ------------------ layer 1 edge scatter: 1 wave / edge, bf16 gather -> f32
__global__ void k_edge1(const long long* ei, const __bf16* g1b, float* agg1) {
  int e = blockIdx.x * 8 + (threadIdx.x >> 5);
  if (e >= N_EDGES) return;
  int j = threadIdx.x & 31;
  int s = (int)ei[e];
  int d = (int)ei[N_EDGES + e];
  bf4 v = *(const bf4*)(g1b + (size_t)s * HID + j * 4);
  float* dst = agg1 + (size_t)d * HID + j * 4;
  atomicAdd(dst + 0, bf2f(v[0]));
  atomicAdd(dst + 1, bf2f(v[1]));
  atomicAdd(dst + 2, bf2f(v[2]));
  atomicAdd(dst + 3, bf2f(v[3]));
}

__global__ void k_mean1(const float* agg1, const float* deg, __bf16* mean1b) {
  long long t = (long long)blockIdx.x * blockDim.x + threadIdx.x;
  if (t >= (long long)N_NODES * HID) return;
  int n = (int)(t >> 7);
  mean1b[t] = f2bf(agg1[t] * (1.0f / fmaxf(deg[n], 1.0f)));
}

// -------------------------------------------------- WMMA fragment builders
// A (16x32 bf16): lane l -> row m=(l&15); k-chunks [(l>>4)*8, +8) and +16.
__device__ __forceinline__ v16bf load_a_frag(const __bf16* A, int row_base,
                                             int kt, int lane) {
  int m  = row_base + (lane & 15);
  int kb = kt * 32 + ((lane >> 4) << 3);
  const __bf16* p = A + (size_t)m * HID + kb;
  bf8 lo = *(const bf8*)p;
  bf8 hi = *(const bf8*)(p + 16);
  v16bf fr;
#pragma unroll
  for (int i = 0; i < 8; ++i) { fr[i] = lo[i]; fr[i + 8] = hi[i]; }
  return fr;
}
// B (32x16 bf16), B[k][n] = W[n][k] with W row-major [out][in]:
// lane l -> col n=(l&15); k-chunks [kt*32 + (l>>4)*8, +8) and +16.
__device__ __forceinline__ v16bf load_b_frag(const __bf16* W, int col_base,
                                             int kt, int lane) {
  int n  = col_base + (lane & 15);
  int kb = kt * 32 + ((lane >> 4) << 3);
  const __bf16* p = W + (size_t)n * HID + kb;
  bf8 lo = *(const bf8*)p;
  bf8 hi = *(const bf8*)(p + 16);
  v16bf fr;
#pragma unroll
  for (int i = 0; i < 8; ++i) { fr[i] = lo[i]; fr[i + 8] = hi[i]; }
  return fr;
}

// ---- layer 1 GEMM: x2 = prelu(mean1@Wl1.T + bl1 + g1@Wr1.T, a1)
//      g2 = h + x2 + x_theta   (stored bf16 for layer-2 gather/GEMM)
__global__ void __launch_bounds__(256)
k_gemm1(const __bf16* mean1b, const __bf16* g1b,
        const __bf16* Wl1b, const __bf16* Wr1b,
        const float* bl1, const float* a1,
        const float* h, const float* x_theta, __bf16* g2b) {
  int lane = threadIdx.x & 31;
  int wave = threadIdx.x >> 5;       // 8 waves -> 8 N-tiles of 16 (HID=128)
  int m0 = blockIdx.x * 16;          // 6250 M-tiles (N_NODES = 6250*16)
  int n0 = wave * 16;
  v8f c1 = {}; v8f c2 = {};
#pragma unroll
  for (int kt = 0; kt < 4; ++kt) {
    v16bf af1 = load_a_frag(mean1b, m0, kt, lane);
    v16bf bf1 = load_b_frag(Wl1b,  n0, kt, lane);
    c1 = __builtin_amdgcn_wmma_f32_16x16x32_bf16(false, af1, false, bf1,
                                                 (short)0, c1, false, false);
    v16bf af2 = load_a_frag(g1b,  m0, kt, lane);
    v16bf bf2 = load_b_frag(Wr1b, n0, kt, lane);
    c2 = __builtin_amdgcn_wmma_f32_16x16x32_bf16(false, af2, false, bf2,
                                                 (short)0, c2, false, false);
  }
  int f = n0 + (lane & 15);
  float bb = bl1[f], aa = a1[f];
#pragma unroll
  for (int r = 0; r < 8; ++r) {
    int m = m0 + r + ((lane >> 4) << 3);
    float v = c1[r] + c2[r] + bb;
    v = (v >= 0.0f) ? v : aa * v;
    size_t idx = (size_t)m * HID + f;
    g2b[idx] = f2bf(h[idx] + v + x_theta[idx]);
  }
}

// ----- layer 2 edge scatter: only edges landing in the returned slice
__global__ void k_edge2(const long long* ei, const __bf16* g2b, float* agg2) {
  int e = blockIdx.x * 8 + (threadIdx.x >> 5);
  if (e >= N_EDGES) return;
  int d = (int)ei[N_EDGES + e];
  if (d >= BATCH) return;
  int s = (int)ei[e];
  int j = threadIdx.x & 31;
  bf4 v = *(const bf4*)(g2b + (size_t)s * HID + j * 4);
  float* dst = agg2 + (size_t)d * HID + j * 4;
  atomicAdd(dst + 0, bf2f(v[0]));
  atomicAdd(dst + 1, bf2f(v[1]));
  atomicAdd(dst + 2, bf2f(v[2]));
  atomicAdd(dst + 3, bf2f(v[3]));
}

__global__ void k_mean2(const float* agg2, const float* deg, __bf16* mean2b) {
  int t = blockIdx.x * blockDim.x + threadIdx.x;
  if (t >= BATCH * HID) return;
  int n = t >> 7;
  mean2b[t] = f2bf(agg2[t] * (1.0f / fmaxf(deg[n], 1.0f)));
}

// ---- layer 2 GEMM: out = mean2@Wl2.T + bl2 + g2@Wr2.T, rows [0,1024)
__global__ void __launch_bounds__(128)
k_gemm2(const __bf16* mean2b, const __bf16* g2b,
        const __bf16* Wl2b, const __bf16* Wr2b,
        const float* bl2, float* out) {
  int lane = threadIdx.x & 31;
  int wave = threadIdx.x >> 5;       // 4 waves -> 4 N-tiles (OUTC=64)
  int m0 = blockIdx.x * 16;          // 64 M-tiles (BATCH = 64*16)
  int n0 = wave * 16;
  v8f c1 = {}; v8f c2 = {};
#pragma unroll
  for (int kt = 0; kt < 4; ++kt) {
    v16bf af1 = load_a_frag(mean2b, m0, kt, lane);
    v16bf bf1 = load_b_frag(Wl2b,  n0, kt, lane);
    c1 = __builtin_amdgcn_wmma_f32_16x16x32_bf16(false, af1, false, bf1,
                                                 (short)0, c1, false, false);
    v16bf af2 = load_a_frag(g2b,  m0, kt, lane);
    v16bf bf2 = load_b_frag(Wr2b, n0, kt, lane);
    c2 = __builtin_amdgcn_wmma_f32_16x16x32_bf16(false, af2, false, bf2,
                                                 (short)0, c2, false, false);
  }
  int f = n0 + (lane & 15);
  float bb = bl2[f];
#pragma unroll
  for (int r = 0; r < 8; ++r) {
    int m = m0 + r + ((lane >> 4) << 3);
    out[(size_t)m * OUTC + f] = c1[r] + c2[r] + bb;
  }
}

// ---------------------------------------------------------------- launcher
extern "C" void kernel_launch(void* const* d_in, const int* in_sizes, int n_in,
                              void* d_out, int out_size, void* d_ws, size_t ws_size,
                              hipStream_t stream) {
  (void)in_sizes; (void)n_in; (void)out_size; (void)ws_size;

  const float*      x   = (const float*)d_in[0];
  const long long*  ei  = (const long long*)d_in[1];
  const float*      W1  = (const float*)d_in[3];
  const float*      Wl0 = (const float*)d_in[4];
  const float*      bl0 = (const float*)d_in[5];
  const float*      Wr0 = (const float*)d_in[6];
  const float*      a0  = (const float*)d_in[7];
  const float*      Wl1 = (const float*)d_in[8];
  const float*      bl1 = (const float*)d_in[9];
  const float*      Wr1 = (const float*)d_in[10];
  const float*      a1  = (const float*)d_in[11];
  const float*      Wl2 = (const float*)d_in[12];
  const float*      bl2 = (const float*)d_in[13];
  const float*      Wr2 = (const float*)d_in[14];
  float*            out = (float*)d_out;

  // workspace carve-out (256B aligned); zeroed region first & contiguous
  char* w = (char*)d_ws;
  auto alloc = [&](size_t bytes) -> void* {
    void* p = (void*)w;
    w += (bytes + 255) & ~(size_t)255;
    return p;
  };
  float*  deg    = (float*)alloc((size_t)N_NODES * 4);
  float*  agg0   = (float*)alloc((size_t)N_NODES * 16);
  float*  agg1   = (float*)alloc((size_t)N_NODES * HID * 4);
  float*  agg2   = (float*)alloc((size_t)BATCH * HID * 4);
  char*   zero_end = w;
  float*  x_theta = (float*)alloc((size_t)N_NODES * HID * 4);
  float*  h       = (float*)alloc((size_t)N_NODES * HID * 4);
  __bf16* g1b    = (__bf16*)alloc((size_t)N_NODES * HID * 2);
  __bf16* mean1b = (__bf16*)alloc((size_t)N_NODES * HID * 2);
  __bf16* g2b    = (__bf16*)alloc((size_t)N_NODES * HID * 2);
  __bf16* mean2b = (__bf16*)alloc((size_t)BATCH * HID * 2);
  __bf16* Wl1b   = (__bf16*)alloc((size_t)HID * HID * 2);
  __bf16* Wr1b   = (__bf16*)alloc((size_t)HID * HID * 2);
  __bf16* Wl2b   = (__bf16*)alloc((size_t)OUTC * HID * 2);
  __bf16* Wr2b   = (__bf16*)alloc((size_t)OUTC * HID * 2);

  size_t zero_floats = (size_t)(zero_end - (char*)deg) / 4;
  k_zero<<<4096, 256, 0, stream>>>(deg, zero_floats);
  k_cvt_w<<<192, 256, 0, stream>>>(Wl1, Wr1, Wl2, Wr2, Wl1b, Wr1b, Wl2b, Wr2b);

  k_edge0<<<(N_EDGES + 255) / 256, 256, 0, stream>>>(ei, x, deg, agg0);
  k_node0<<<(N_NODES * HID + 255) / 256, 256, 0, stream>>>(
      x, deg, agg0, W1, Wl0, bl0, Wr0, a0, x_theta, h, g1b);

  k_edge1<<<(N_EDGES + 7) / 8, 256, 0, stream>>>(ei, g1b, agg1);
  k_mean1<<<(N_NODES * HID + 255) / 256, 256, 0, stream>>>(agg1, deg, mean1b);
  k_gemm1<<<N_NODES / 16, 256, 0, stream>>>(mean1b, g1b, Wl1b, Wr1b,
                                            bl1, a1, h, x_theta, g2b);

  k_edge2<<<(N_EDGES + 7) / 8, 256, 0, stream>>>(ei, g2b, agg2);
  k_mean2<<<(BATCH * HID + 255) / 256, 256, 0, stream>>>(agg2, deg, mean2b);
  k_gemm2<<<BATCH / 16, 128, 0, stream>>>(mean2b, g2b, Wl2b, Wr2b, bl2, out);
}
